// SGCN_10737418240768
// MI455X (gfx1250) — matrix operand order repaired and checked by reading, not compiled, not run on hardware
//
#include <hip/hip_runtime.h>
#include <hip/hip_bf16.h>
#include <math.h>

// Problem constants (match reference)
#define Bsz   64
#define Hdim  16384
#define BS    32
#define NB    512                 // Hdim / BS
#define NNZ   26214
#define N_IN  1024
#define N_OUT 1024
#define STEPS 8

typedef __attribute__((ext_vector_type(16))) __bf16 v16bf;
typedef __attribute__((ext_vector_type(8)))  __bf16 v8bf;
typedef __attribute__((ext_vector_type(8)))  float  v8f;

static __device__ __forceinline__ float sigmoidf_(float x) {
    return 1.0f / (1.0f + __expf(-x));
}

// ---------------------------------------------------------------------------
// Phase 0a: initial activations  hs[b,h] = (h < N_IN) ? inp[b,h] : 0   (bf16)
// ---------------------------------------------------------------------------
__global__ void k_init_hs(const float* __restrict__ inp, __bf16* __restrict__ hs) {
    int idx = blockIdx.x * blockDim.x + threadIdx.x;
    if (idx >= Bsz * Hdim) return;
    int b = idx / Hdim, h = idx - b * Hdim;
    hs[idx] = (h < N_IN) ? (__bf16)inp[b * N_IN + h] : (__bf16)0.0f;
}

// ---------------------------------------------------------------------------
// Phase 0b: CSR over block_rows.  count -> scan -> scatter -> per-row sort
// (insertion sort makes in-row order deterministic: ascending block id)
// ---------------------------------------------------------------------------
__global__ void k_count(const int* __restrict__ rows, int* __restrict__ cnt) {
    int n = blockIdx.x * blockDim.x + threadIdx.x;
    if (n < NNZ) atomicAdd(&cnt[rows[n]], 1);
}

__global__ void k_scan(const int* __restrict__ cnt, int* __restrict__ rowStart) {
    __shared__ int sh[NB];
    int t = threadIdx.x;
    sh[t] = cnt[t];
    __syncthreads();
    for (int off = 1; off < NB; off <<= 1) {
        int v = (t >= off) ? sh[t - off] : 0;
        __syncthreads();
        sh[t] += v;
        __syncthreads();
    }
    rowStart[t + 1] = sh[t];
    if (t == 0) rowStart[0] = 0;
}

__global__ void k_scatter(const int* __restrict__ rows, const int* __restrict__ rowStart,
                          int* __restrict__ cursor, int* __restrict__ rowPerm) {
    int n = blockIdx.x * blockDim.x + threadIdx.x;
    if (n >= NNZ) return;
    int r = rows[n];
    int pos = atomicAdd(&cursor[r], 1);
    rowPerm[rowStart[r] + pos] = n;
}

__global__ void k_sortrows(int* __restrict__ rowPerm, const int* __restrict__ rowStart) {
    int r = blockIdx.x * blockDim.x + threadIdx.x;
    if (r >= NB) return;
    int s0 = rowStart[r], s1 = rowStart[r + 1];
    for (int i = s0 + 1; i < s1; ++i) {
        int key = rowPerm[i];
        int j = i - 1;
        while (j >= s0 && rowPerm[j] > key) { rowPerm[j + 1] = rowPerm[j]; --j; }
        rowPerm[j + 1] = key;
    }
}

// Pre-gather the column index stream in sorted order: one scalar load per
// WMMA in the hot loop instead of a 2-deep dependent s_load chain.
__global__ void k_gather_cols(const int* __restrict__ rowPerm,
                              const int* __restrict__ blockCols,
                              int* __restrict__ colSorted) {
    int s = blockIdx.x * blockDim.x + threadIdx.x;
    if (s < NNZ) colSorted[s] = blockCols[rowPerm[s]];
}

// ---------------------------------------------------------------------------
// Phase 0c: convert fp32 32x32 blocks -> bf16, pre-swizzled into the exact
// V_WMMA_F32_16X16X32_BF16 B-operand register layout (ISA 7.12.2) AND stored
// in row-sorted order, so the hot loop's weight stream is fully contiguous:
//   wfrag[s*1024 + nTile*512 + lane*16 + kk]
//     = W[rowPerm[s]][nTile*16 + (lane&15), (lane>>4)*8 + (kk&7) + (kk>>3)*16]
// (B is K(32) x N(16); lane l holds N = l&15, K = (l>>4)*8 + (kk&7) + (kk>>3)*16.)
// ---------------------------------------------------------------------------
__global__ void k_conv_blocks(const float* __restrict__ blocks,
                              const int* __restrict__ rowPerm,
                              __bf16* __restrict__ wfrag) {
    long long idx = (long long)blockIdx.x * blockDim.x + threadIdx.x;
    if (idx >= (long long)NNZ * 1024) return;
    int s = (int)(idx >> 10);
    int o = (int)(idx & 1023);
    int n = rowPerm[s];
    int nTile = o >> 9;
    int rem   = o & 511;
    int lane  = rem >> 4;
    int kk    = rem & 15;
    int i = nTile * 16 + (lane & 15);   // output index within block (N)
    int j = (lane >> 4) * 8 + (kk & 7) + (kk >> 3) * 16;      // k index within block (K)
    wfrag[idx] = (__bf16)blocks[(long long)n * 1024 + i * 32 + j];
}

// ---------------------------------------------------------------------------
// Fragment loader for the hot loop.
//   A fragment (16x32 bf16, ISA layout): lane l holds M = l&15,
//     K = {kb..kb+7, kb+16..kb+23}, kb = (l>=16)*8 -> two contiguous b128 loads
//   B fragment: pre-swizzled + sorted -> one contiguous 32B load at idx*2KB.
// ---------------------------------------------------------------------------
static __device__ __forceinline__ void
load_frag_c(const __bf16* __restrict__ aRow, int kBase, int c,
            const __bf16* __restrict__ wbase, int idx, v16bf& a, v16bf& b) {
    const __bf16* xp = aRow + c * BS + kBase;
    union { v16bf v; v8bf h[2]; } u;
    u.h[0] = *(const v8bf*)(xp);
    u.h[1] = *(const v8bf*)(xp + 16);
    a = u.v;
    b = *(const v16bf*)(wbase + (long long)idx * 1024);
}

static __device__ __forceinline__ v8f
wmma_bf16(v16bf a, v16bf b, v8f c) {
    return __builtin_amdgcn_wmma_f32_16x16x32_bf16(
        /*neg_a=*/false, a, /*neg_b=*/false, b,
        /*c_mod=*/(short)0, c, /*reuse_a=*/false, /*reuse_b=*/false);
}

// WGP-scope prefetch into the near cache (builtin only reaches SE/SYS scope).
static __device__ __forceinline__ void prefetch_wgp(const void* p) {
    asm volatile("global_prefetch_b8 %0, off" :: "v"(p) : "memory");
}

// ---------------------------------------------------------------------------
// Main step: block-sparse SpMM via WMMA.
//   grid = NB row-blocks, 256 threads = 8 wave32.
//   wave wid -> (mTile = wid>>1 over batch(64), nTile = wid&1 over out cols(32))
//   Single-basic-block software pipeline: both register buffers are reloaded
//   every iteration with clamped indices (no conditional load path), so the
//   backend can emit partial s_wait_loadcnt and overlap VMEM with the WMMA
//   pipe.  Weight stream is contiguous; WGP-scope prefetch runs 8KB ahead.
//   Epilogue (C/D layout): VGPR v, lane l -> M = v + 8*(l>=16), N = l&15.
// ---------------------------------------------------------------------------
__global__ void __launch_bounds__(256)
k_spmm(const __bf16* __restrict__ hsIn,
       const __bf16* __restrict__ wfrag,     // sorted B-fragments
       const int*    __restrict__ rowStart,
       const int*    __restrict__ colSorted, // sorted column indices
       const float*  __restrict__ bias,
       __bf16*       __restrict__ hsOut,
       float*        __restrict__ out,
       int lastStep)
{
    const int r     = blockIdx.x;          // output row-block 0..NB-1
    const int wid   = threadIdx.x >> 5;    // wave 0..7
    const int lane  = threadIdx.x & 31;
    const int mTile = wid >> 1;            // batch tile 0..3
    const int nTile = wid & 1;             // out-col tile 0..1

    v8f acc = {};
    const int s0  = rowStart[r];
    const int cnt = rowStart[r + 1] - s0;

    const int mRow  = mTile * 16 + (lane & 15);   // batch row feeding A
    const int kBase = (lane >> 4) * 8;            // A-fragment K base
    const __bf16* __restrict__ aRow  = hsIn + (long long)mRow * Hdim;
    const __bf16* __restrict__ wbase = wfrag + (long long)s0 * 1024 + nTile * 512 + lane * 16;
    const int* __restrict__ colS = colSorted + s0;

    if (cnt > 0) {
        const int cLast = cnt - 1;
        v16bf a0, b0, a1, b1;
        {
            int j1 = (1 <= cLast) ? 1 : cLast;
            int c0 = colS[0];
            int c1 = colS[j1];
            load_frag_c(aRow, kBase, c0, wbase, 0,  a0, b0);
            load_frag_c(aRow, kBase, c1, wbase, j1, a1, b1);
        }
        int i = 0;
        for (; i + 2 <= cnt; i += 2) {
            // next-pair indices, clamped -> no conditional load path
            int j2 = (i + 2 <= cLast) ? i + 2 : cLast;
            int j3 = (i + 3 <= cLast) ? i + 3 : cLast;
            int jp = (i + 4 <= cLast) ? i + 4 : cLast;
            int c2 = colS[j2];                       // scalar loads issued early
            int c3 = colS[j3];
            prefetch_wgp(wbase + (long long)jp * 1024);  // 8KB ahead, WGP scope
            acc = wmma_bf16(a0, b0, acc);
            load_frag_c(aRow, kBase, c2, wbase, j2, a0, b0);
            acc = wmma_bf16(a1, b1, acc);
            load_frag_c(aRow, kBase, c3, wbase, j3, a1, b1);
        }
        if (i < cnt) acc = wmma_bf16(a0, b0, acc);   // odd tail
    }

    // Epilogue: bias + sigmoid, write bf16 next-state or f32 final output
    const int hiHalf = (lane >> 4) ? 8 : 0;
    const int col    = r * BS + nTile * 16 + (lane & 15);
    const float bv   = bias[col];
#pragma unroll
    for (int v = 0; v < 8; ++v) {
        int bRow = mTile * 16 + v + hiHalf;
        float y  = acc[v] + bv;
        if (!lastStep) {
            hsOut[(long long)bRow * Hdim + col] = (__bf16)sigmoidf_(y);
        } else if (col >= Hdim - N_OUT) {
            out[(long long)bRow * N_OUT + (col - (Hdim - N_OUT))] = sigmoidf_(y);
        }
    }
}

// ---------------------------------------------------------------------------
// Host launcher
// ---------------------------------------------------------------------------
extern "C" void kernel_launch(void* const* d_in, const int* in_sizes, int n_in,
                              void* d_out, int out_size, void* d_ws, size_t ws_size,
                              hipStream_t stream) {
    const float* inp       = (const float*)d_in[0];   // [B, N_IN]
    const float* blocks    = (const float*)d_in[1];   // [NNZ, 32, 32]
    const float* bias      = (const float*)d_in[2];   // [H]
    const int*   blockRows = (const int*)  d_in[3];   // [NNZ]
    const int*   blockCols = (const int*)  d_in[4];   // [NNZ]
    float*       out       = (float*)d_out;           // [B, N_OUT]

    // Workspace layout (256B aligned), ~58 MB total:
    char* ws = (char*)d_ws;
    size_t off = 0;
    auto take = [&](size_t bytes) { void* p = ws + off; off += (bytes + 255) & ~(size_t)255; return p; };
    __bf16* wfrag     = (__bf16*)take((size_t)NNZ * 1024 * 2);  // 53.7 MB sorted bf16 B-fragments
    __bf16* hsA       = (__bf16*)take((size_t)Bsz * Hdim * 2);  // 2 MB
    __bf16* hsB       = (__bf16*)take((size_t)Bsz * Hdim * 2);  // 2 MB
    int*    cnt       = (int*)   take(NB * 4);
    int*    rowStart  = (int*)   take((NB + 1) * 4);
    int*    cursor    = (int*)   take(NB * 4);
    int*    rowPerm   = (int*)   take((size_t)NNZ * 4);
    int*    colSorted = (int*)   take((size_t)NNZ * 4);
    (void)off; (void)ws_size;

    // --- Phase 0: CSR build + weight conversion (all graph-capturable) ---
    hipMemsetAsync(cnt,    0, NB * 4, stream);
    hipMemsetAsync(cursor, 0, NB * 4, stream);

    {
        int n = Bsz * Hdim;
        k_init_hs<<<(n + 255) / 256, 256, 0, stream>>>(inp, hsA);
    }
    k_count  <<<(NNZ + 255) / 256, 256, 0, stream>>>(blockRows, cnt);
    k_scan   <<<1, NB, 0, stream>>>(cnt, rowStart);
    k_scatter<<<(NNZ + 255) / 256, 256, 0, stream>>>(blockRows, rowStart, cursor, rowPerm);
    k_sortrows<<<(NB + 255) / 256, 256, 0, stream>>>(rowPerm, rowStart);
    k_gather_cols<<<(NNZ + 255) / 256, 256, 0, stream>>>(rowPerm, blockCols, colSorted);
    {
        long long n = (long long)NNZ * 1024;
        k_conv_blocks<<<(unsigned)((n + 255) / 256), 256, 0, stream>>>(blocks, rowPerm, wfrag);
    }

    // --- 8 block-sparse WMMA steps, ping-pong bf16 activations ---
    __bf16* cur = hsA;
    __bf16* nxt = hsB;
    for (int step = 0; step < STEPS; ++step) {
        int last = (step == STEPS - 1);
        k_spmm<<<NB, 256, 0, stream>>>(cur, wfrag, rowStart, colSorted,
                                       bias, nxt, out, last);
        __bf16* t = cur; cur = nxt; nxt = t;
    }
}
